// VectorQuantizer_69131793596563
// MI455X (gfx1250) — matrix operand order repaired and checked by reading, not compiled, not run on hardware
//
#include <hip/hip_runtime.h>

#define N_TOK 32768
#define DIM   256
#define KCB   8192
#define LDS_PITCH 264            // 256 halves + 8 pad -> 528B rows, conflict-free b128
#define PLANE_HALF (64 * LDS_PITCH)      // one hi or lo plane, in halves
#define BUF_HALF   (2 * PLANE_HALF)      // hi+lo buffer, in halves
#define NSTAGES    (KCB / 64)            // 128

typedef __attribute__((ext_vector_type(16))) __bf16 v16bf;
typedef __attribute__((ext_vector_type(8)))  float  v8f;

union Frag {
    v16bf v;
    float4 f4[2];
    unsigned short u[16];
};

__device__ __forceinline__ unsigned short f32_to_bf16(float f) {
    unsigned int u = __float_as_uint(f);
    u += 0x7fffu + ((u >> 16) & 1u);   // round-to-nearest-even
    return (unsigned short)(u >> 16);
}
__device__ __forceinline__ float bf16_to_f32(unsigned short h) {
    return __uint_as_float(((unsigned int)h) << 16);
}

// Async global->LDS copy of 16 bytes/lane (CDNA5, tracked by ASYNCcnt).
// ISA 10.2: flat shared addresses carry the LDS byte offset in addr[31:0]
// (aperture tag is addr[63:32]), so the truncated pointer is both the correct
// VDST operand AND a genuine ptrtoint escape of the LDS object, which keeps
// alias analysis from proving the LDS "never written" and folding its loads.
__device__ __forceinline__ void async_copy_b128(unsigned short* lds_dst, const void* gsrc) {
    asm volatile("global_load_async_to_lds_b128 %0, %1, off"
                 :: "v"((unsigned)(unsigned long long)lds_dst), "v"(gsrc)
                 : "memory");
}

// ---------------------------------------------------------------- prep z_e
__global__ void vq_prep_z(const float* __restrict__ z,
                          unsigned short* __restrict__ zh,
                          unsigned short* __restrict__ zl) {
    size_t i = (size_t)blockIdx.x * blockDim.x + threadIdx.x;
    float v = z[i];
    unsigned short h = f32_to_bf16(v);
    zh[i] = h;
    zl[i] = f32_to_bf16(v - bf16_to_f32(h));
}

// --------------------------------------------------- prep codebook + norms
__global__ void vq_prep_e(const float* __restrict__ cbk,
                          unsigned short* __restrict__ eh,
                          unsigned short* __restrict__ el,
                          float* __restrict__ en2) {
    __shared__ float red[256];
    int d = threadIdx.x;
    size_t i = (size_t)blockIdx.x * DIM + d;
    float v = cbk[i];
    unsigned short h = f32_to_bf16(v);
    eh[i] = h;
    el[i] = f32_to_bf16(v - bf16_to_f32(h));
    red[d] = v * v;
    __syncthreads();
    for (int s = 128; s > 0; s >>= 1) {
        if (d < s) red[d] += red[d + s];
        __syncthreads();
    }
    if (d == 0) en2[blockIdx.x] = red[0];
}

__global__ void vq_zero(float* loss) { *loss = 0.0f; }

// ------------------------------------------------ fused distance + argmin
// Block: 256 threads = 8 waves, each wave owns 16 rows of z (A kept in regs).
// Codebook staged 64 rows/stage (hi+lo) into a ping-pong LDS double buffer
// filled by global_load_async_to_lds_b128 overlapped with WMMA compute.
__global__ __launch_bounds__(256)
__attribute__((amdgpu_waves_per_eu(1, 4)))
void vq_argmin(const unsigned short* __restrict__ zh,
               const unsigned short* __restrict__ zl,
               const unsigned short* __restrict__ eh,
               const unsigned short* __restrict__ el,
               const float* __restrict__ en2,
               int* __restrict__ out_idx) {
    // [buf0: hi | lo][buf1: hi | lo]
    __shared__ unsigned short sB[2 * BUF_HALF];

    const int tid  = threadIdx.x;
    const int wave = tid >> 5;
    const int lane = tid & 31;
    const int nl   = lane & 15;   // M for A-frag, N for B/C-frag
    const int g    = lane >> 4;   // half-wave group

    const int row0 = blockIdx.x * 128 + wave * 16;
    const int arow = row0 + nl;

    // Preload full A operand: 16 rows x 256 K, hi+lo. A layout (matches B):
    // VGPR0-3 hold K = g*16 + 0..7, VGPR4-7 hold K = g*16 + 8..15.
    Frag ah[8], al[8];
#pragma unroll
    for (int c = 0; c < 8; ++c) {
        const size_t base = (size_t)arow * DIM + c * 32;
        ah[c].f4[0] = *(const float4*)&zh[base + g * 16];
        ah[c].f4[1] = *(const float4*)&zh[base + g * 16 + 8];
        al[c].f4[0] = *(const float4*)&zl[base + g * 16];
        al[c].f4[1] = *(const float4*)&zl[base + g * 16 + 8];
    }

    float minv[8];
    int   mini[8];
#pragma unroll
    for (int r = 0; r < 8; ++r) { minv[r] = 3.4e38f; mini[r] = 0; }

    // staging role: 16 async b128 per wave per stage (8 hi + 8 lo)
    const int srow = tid >> 2;                 // 0..63 codebook row in stage
    const int sseg = tid & 3;                  // quarter row (64 halves = 128B)
    const int lds_row = srow * LDS_PITCH + sseg * 64;   // halves

    // prologue: kick off stage 0 into buffer 0
    {
        const size_t gbase = (size_t)srow * DIM + sseg * 64;
#pragma unroll
        for (int q = 0; q < 8; ++q) {
            async_copy_b128(&sB[lds_row + q * 8],              (const char*)&eh[gbase] + q * 16);
            async_copy_b128(&sB[lds_row + PLANE_HALF + q * 8], (const char*)&el[gbase] + q * 16);
        }
    }

    for (int stage = 0; stage < NSTAGES; ++stage) {
        const int p = stage & 1;
        if (stage + 1 < NSTAGES) {
            // issue next stage into the other buffer, then wait for ours:
            // async loads complete in order, so ASYNCcnt<=16 => ours landed.
            unsigned short* nb = &sB[(p ^ 1) * BUF_HALF + lds_row];
            const size_t gbase = (size_t)((stage + 1) * 64 + srow) * DIM + sseg * 64;
#pragma unroll
            for (int q = 0; q < 8; ++q) {
                async_copy_b128(nb + q * 8,              (const char*)&eh[gbase] + q * 16);
                async_copy_b128(nb + PLANE_HALF + q * 8, (const char*)&el[gbase] + q * 16);
            }
            asm volatile("s_wait_asynccnt 16" ::: "memory");
        } else {
            asm volatile("s_wait_asynccnt 0" ::: "memory");
        }
        __syncthreads();   // every wave's view of buf[p] is complete

        const unsigned short* sHi = &sB[p * BUF_HALF];
        const unsigned short* sLo = sHi + PLANE_HALF;
        // opaque use of the read pointers: loads below must re-read LDS that
        // the async engine (not visible to the compiler) just filled.
        asm volatile("" :: "s"(sHi), "s"(sLo) : "memory");

#pragma unroll
        for (int sub = 0; sub < 4; ++sub) {
            const int cb0 = stage * 64 + sub * 16;
            v8f acc = {};
#pragma unroll
            for (int c = 0; c < 8; ++c) {
                // B as loaded here (same k order as A): lane n = nl, halves = K = g*16 + 0..15
                Frag bh, bl;
                const float4* pb = (const float4*)&sHi[(sub * 16 + nl) * LDS_PITCH + c * 32 + g * 16];
                bh.f4[0] = pb[0]; bh.f4[1] = pb[1];
                const float4* pl = (const float4*)&sLo[(sub * 16 + nl) * LDS_PITCH + c * 32 + g * 16];
                bl.f4[0] = pl[0]; bl.f4[1] = pl[1];
                // fp32-accurate dot via bf16x3: hi*hi + hi*lo + lo*hi
                acc = __builtin_amdgcn_wmma_f32_16x16x32_bf16(
                    false, ah[c].v, false, bh.v, (short)0, acc, false, false);
                acc = __builtin_amdgcn_wmma_f32_16x16x32_bf16(
                    false, ah[c].v, false, bl.v, (short)0, acc, false, false);
                acc = __builtin_amdgcn_wmma_f32_16x16x32_bf16(
                    false, al[c].v, false, bh.v, (short)0, acc, false, false);
            }
            const float cn2  = en2[cb0 + nl];
            const int   cidx = cb0 + nl;
#pragma unroll
            for (int r = 0; r < 8; ++r) {
                float d = cn2 - 2.0f * acc[r];   // ||z||^2 dropped (row-constant)
                if (d < minv[r]) { minv[r] = d; mini[r] = cidx; }
            }
        }
        __syncthreads();   // all waves done reading buf[p] before stage+2 refills it
    }

    // min-reduce across the 16 lanes of each half-wave (tie -> lowest index)
#pragma unroll
    for (int r = 0; r < 8; ++r) {
#pragma unroll
        for (int off = 1; off < 16; off <<= 1) {
            float ov = __shfl_xor(minv[r], off, 32);
            int   oi = __shfl_xor(mini[r], off, 32);
            if (ov < minv[r] || (ov == minv[r] && oi < mini[r])) {
                minv[r] = ov; mini[r] = oi;
            }
        }
        if (nl == 0) out_idx[row0 + g * 8 + r] = mini[r];
    }
}

// ------------------------------------- gather z_q, loss, outputs
__global__ void vq_finalize(const float* __restrict__ z,
                            const float* __restrict__ cbk,
                            const int* __restrict__ idx,
                            float* __restrict__ out_zq,
                            float* __restrict__ out_loss,
                            float* __restrict__ out_idx_f) {
    __shared__ float red[256];
    const int r = blockIdx.x;
    const int d = threadIdx.x;
    const int k = idx[r];
    const float ze = z[(size_t)r * DIM + d];
    const float zq = cbk[(size_t)k * DIM + d];
    out_zq[(size_t)r * DIM + d] = ze + (zq - ze);   // straight-through, as written
    const float diff = zq - ze;
    red[d] = diff * diff;
    __syncthreads();
    for (int s = 128; s > 0; s >>= 1) {
        if (d < s) red[d] += red[d + s];
        __syncthreads();
    }
    if (d == 0) {
        // codebook_loss + 0.25*commitment_loss = 1.25 * mean(diff^2)
        atomicAdd(out_loss, red[0] * (1.25f / ((float)N_TOK * (float)DIM)));
        out_idx_f[r] = (float)k;
    }
}

extern "C" void kernel_launch(void* const* d_in, const int* in_sizes, int n_in,
                              void* d_out, int out_size, void* d_ws, size_t ws_size,
                              hipStream_t stream) {
    const float* z   = (const float*)d_in[0];
    const float* cbk = (const float*)d_in[1];

    float* out      = (float*)d_out;
    float* out_zq   = out;                          // N*D
    float* out_loss = out + (size_t)N_TOK * DIM;    // 1
    float* out_idx  = out_loss + 1;                 // N (indices as float)

    char* ws = (char*)d_ws;
    unsigned short* zh = (unsigned short*)ws; ws += (size_t)N_TOK * DIM * 2;
    unsigned short* zl = (unsigned short*)ws; ws += (size_t)N_TOK * DIM * 2;
    unsigned short* eh = (unsigned short*)ws; ws += (size_t)KCB * DIM * 2;
    unsigned short* el = (unsigned short*)ws; ws += (size_t)KCB * DIM * 2;
    float* en2 = (float*)ws;                  ws += (size_t)KCB * 4;
    int*   idx = (int*)ws;

    vq_prep_z<<<(N_TOK * DIM) / 256, 256, 0, stream>>>(z, zh, zl);
    vq_prep_e<<<KCB, 256, 0, stream>>>(cbk, eh, el, en2);
    vq_zero<<<1, 1, 0, stream>>>(out_loss);
    vq_argmin<<<N_TOK / 128, 256, 0, stream>>>(zh, zl, eh, el, en2, idx);
    vq_finalize<<<N_TOK, 256, 0, stream>>>(z, cbk, idx, out_zq, out_loss, out_idx);
}
